// GAT_66709432041919
// MI455X (gfx1250) — matrix operand order repaired and checked by reading, not compiled, not run on hardware
//
#include <hip/hip_runtime.h>
#include <hip/hip_bf16.h>
#include <math.h>

typedef __attribute__((ext_vector_type(16))) __bf16 v16bf;
typedef __attribute__((ext_vector_type(8)))  float  v8f;

#define IN_DIM 128
#define HID    64
#define TPB    256

// ---------------------------------------------------------------- utilities
__device__ __forceinline__ void atomicMaxF(float* addr, float val) {
    // sign-aware integer punning; emits global_atomic_{max_i32,min_u32}
    if (val >= 0.0f) atomicMax((int*)addr, __float_as_int(val));
    else             atomicMin((unsigned int*)addr, __float_as_uint(val));
}

__device__ __forceinline__ float lrelu(float v) { return v > 0.0f ? v : 0.2f * v; }

// ---------------------------------------------------------------- init
__global__ void gat_init_kernel(float* m1, float* s1, float* agg1,
                                float* m2, float* s2, float* agg2, int n) {
    int i = blockIdx.x * blockDim.x + threadIdx.x;
    if (i < n * HID) agg1[i] = 0.0f;
    if (i < n * 2)   agg2[i] = 0.0f;
    if (i < n) {
        m1[i] = -INFINITY; s1[i] = 0.0f;
        m2[i] = -INFINITY; s2[i] = 0.0f;
    }
}

// ---------------------------------------------------------------- layer-1 GEMM (WMMA bf16)
// xl = x@Wl + bl, xr = x@Wr + br    x:[n,128] f32, W:[128,64], out:[n,64] f32
// One wave computes a 16-row x 64-col tile of both outputs.
__global__ __launch_bounds__(TPB) void gat_gemm1_wmma(
    const float* __restrict__ x,
    const float* __restrict__ Wl, const float* __restrict__ bl,
    const float* __restrict__ Wr, const float* __restrict__ br,
    float* __restrict__ xl, float* __restrict__ xr, int n)
{
    // weights in LDS, bf16, column-major: sW[col*128 + k]
    __shared__ __bf16 sWl[HID * IN_DIM];
    __shared__ __bf16 sWr[HID * IN_DIM];
    for (int t = threadIdx.x; t < HID * IN_DIM; t += TPB) {
        int col = t >> 7, k = t & (IN_DIM - 1);
        sWl[t] = (__bf16)Wl[k * HID + col];
        sWr[t] = (__bf16)Wr[k * HID + col];
    }
    __syncthreads();

    const int lane  = threadIdx.x & 31;
    const int wave  = threadIdx.x >> 5;
    const int row0  = blockIdx.x * (8 * 16) + wave * 16;   // 8 waves * 16 rows
    const int lrow  = lane & 15;
    const int half8 = (lane >> 4) << 3;   // 0 or 8
    const int ahalf = half8;              // A frag: K offset (ahalf<<1) = 0 or 16
    const int bhalf = half8 << 1;         // B frag: K offset 0 or 16

    // ---- A fragments (16x32 bf16 per k-step), contiguous k = 16*(lane>>4) + e
    const int  arow  = row0 + lrow;
    const bool rowok = arow < n;
    const float* xrow = x + (size_t)(rowok ? arow : 0) * IN_DIM;
    v16bf afrag[4];
#pragma unroll
    for (int ks = 0; ks < 4; ++ks) {
#pragma unroll
        for (int e = 0; e < 16; ++e) {
            int k = ks * 32 + (ahalf << 1) + e;
            afrag[ks][e] = rowok ? (__bf16)xrow[k] : (__bf16)0.0f;
        }
    }

    const bool fullTile = (row0 + 16) <= n;   // wave-uniform -> scalar branch

    // ---- 4 column tiles of 16
#pragma unroll
    for (int nt = 0; nt < 4; ++nt) {
        const int col = nt * 16 + lrow;
        v8f accL = {}; v8f accR = {};
#pragma unroll
        for (int ks = 0; ks < 4; ++ks) {
            v16bf bfl, bfr;                 // B frag: 32x16, half-K per lane group
#pragma unroll
            for (int e = 0; e < 16; ++e) {
                int k = ks * 32 + bhalf + e;
                bfl[e] = sWl[col * IN_DIM + k];
                bfr[e] = sWr[col * IN_DIM + k];
            }
            accL = __builtin_amdgcn_wmma_f32_16x16x32_bf16(
                false, afrag[ks], false, bfl, (short)0, accL, false, false);
            accR = __builtin_amdgcn_wmma_f32_16x16x32_bf16(
                false, afrag[ks], false, bfr, (short)0, accR, false, false);
        }
        const float biasL = bl[col], biasR = br[col];
        // C/D layout: VGPR v holds row M = v + 8*(lane>>4), lane holds column col
        float* pL = xl + (size_t)(row0 + half8) * HID + col;
        float* pR = xr + (size_t)(row0 + half8) * HID + col;
        if (fullTile) {
#pragma unroll
            for (int v = 0; v < 8; ++v) {
                pL[(size_t)v * HID] = accL[v] + biasL;
                pR[(size_t)v * HID] = accR[v] + biasR;
            }
        } else {
#pragma unroll
            for (int v = 0; v < 8; ++v) {
                int m = row0 + half8 + v;
                if (m < n) {
                    pL[(size_t)v * HID] = accL[v] + biasL;
                    pR[(size_t)v * HID] = accR[v] + biasR;
                }
            }
        }
    }
}

// ---------------------------------------------------------------- layer-1 edge logits + segment max
__global__ void gat_logit1(const int* __restrict__ ei, int E, int n,
                           const float* __restrict__ xl, const float* __restrict__ xr,
                           const float* __restrict__ att,
                           float* __restrict__ elog, float* __restrict__ m)
{
    int i = blockIdx.x * blockDim.x + threadIdx.x;
    int EP = E + n;
    if (i >= EP) return;
    int src, dst;
    if (i < E) { src = ei[i]; dst = ei[E + i]; } else { src = dst = i - E; }
    const float4* a  = (const float4*)(xl + (size_t)src * HID);
    const float4* b  = (const float4*)(xr + (size_t)dst * HID);
    const float4* at = (const float4*)att;
    float acc = 0.0f;
#pragma unroll
    for (int j = 0; j < HID / 4; ++j) {
        float4 va = a[j], vb = b[j], vt = at[j];
        acc += lrelu(va.x + vb.x) * vt.x;
        acc += lrelu(va.y + vb.y) * vt.y;
        acc += lrelu(va.z + vb.z) * vt.z;
        acc += lrelu(va.w + vb.w) * vt.w;
    }
    elog[i] = acc;
    atomicMaxF(&m[dst], acc);
}

// ---------------------------------------------------------------- exp + segment sum (shared by both layers)
__global__ void gat_expsum(const int* __restrict__ ei, int E, int n,
                           const float* __restrict__ m,
                           float* __restrict__ ep, float* __restrict__ s)
{
    int i = blockIdx.x * blockDim.x + threadIdx.x;
    int EP = E + n;
    if (i >= EP) return;
    int dst = (i < E) ? ei[E + i] : (i - E);
    float p = __expf(ep[i] - m[dst]);
    ep[i] = p;
    atomicAdd(&s[dst], p);
}

// ---------------------------------------------------------------- layer-1 scatter aggregate (4 threads/edge)
__global__ void gat_agg1(const int* __restrict__ ei, int E, int n,
                         const float* __restrict__ p, const float* __restrict__ s,
                         const float* __restrict__ xl, float* __restrict__ agg)
{
    int i = blockIdx.x * blockDim.x + threadIdx.x;
    int edge = i >> 2;
    int EP = E + n;
    if (edge >= EP) return;
    int c = (i & 3) * 16;
    int src, dst;
    if (edge < E) { src = ei[edge]; dst = ei[E + edge]; } else { src = dst = edge - E; }
    float alpha = p[edge] / s[dst];
    const float4* a = (const float4*)(xl + (size_t)src * HID + c);
    float* o = agg + (size_t)dst * HID + c;
#pragma unroll
    for (int j = 0; j < 4; ++j) {
        float4 v = a[j];
        atomicAdd(o + j * 4 + 0, alpha * v.x);
        atomicAdd(o + j * 4 + 1, alpha * v.y);
        atomicAdd(o + j * 4 + 2, alpha * v.z);
        atomicAdd(o + j * 4 + 3, alpha * v.w);
    }
}

// ---------------------------------------------------------------- ELU + layer-2 linears (64 -> 2)
__global__ void gat_node_l2(const float* __restrict__ agg1, const float* __restrict__ b1,
                            const float* __restrict__ Wl2, const float* __restrict__ bl2,
                            const float* __restrict__ Wr2, const float* __restrict__ br2,
                            float* __restrict__ xl2, float* __restrict__ xr2, int n)
{
    int nd = blockIdx.x * blockDim.x + threadIdx.x;
    if (nd >= n) return;
    const float* g = agg1 + (size_t)nd * HID;
    float l0 = bl2[0], l1 = bl2[1], r0 = br2[0], r1 = br2[1];
#pragma unroll
    for (int d = 0; d < HID; ++d) {
        float v = g[d] + b1[d];
        float h = v > 0.0f ? v : (__expf(v) - 1.0f);   // ELU
        l0 += h * Wl2[2 * d];  l1 += h * Wl2[2 * d + 1];
        r0 += h * Wr2[2 * d];  r1 += h * Wr2[2 * d + 1];
    }
    xl2[2 * nd] = l0; xl2[2 * nd + 1] = l1;
    xr2[2 * nd] = r0; xr2[2 * nd + 1] = r1;
}

// ---------------------------------------------------------------- layer-2 edge logits + segment max
__global__ void gat_logit2(const int* __restrict__ ei, int E, int n,
                           const float* __restrict__ xl2, const float* __restrict__ xr2,
                           const float* __restrict__ att2,
                           float* __restrict__ elog, float* __restrict__ m)
{
    int i = blockIdx.x * blockDim.x + threadIdx.x;
    int EP = E + n;
    if (i >= EP) return;
    int src, dst;
    if (i < E) { src = ei[i]; dst = ei[E + i]; } else { src = dst = i - E; }
    float hx = xl2[2 * src]     + xr2[2 * dst];
    float hy = xl2[2 * src + 1] + xr2[2 * dst + 1];
    float e  = att2[0] * lrelu(hx) + att2[1] * lrelu(hy);
    elog[i] = e;
    atomicMaxF(&m[dst], e);
}

// ---------------------------------------------------------------- layer-2 scatter aggregate
__global__ void gat_agg2(const int* __restrict__ ei, int E, int n,
                         const float* __restrict__ p, const float* __restrict__ s,
                         const float* __restrict__ xl2, float* __restrict__ agg2)
{
    int i = blockIdx.x * blockDim.x + threadIdx.x;
    int EP = E + n;
    if (i >= EP) return;
    int src, dst;
    if (i < E) { src = ei[i]; dst = ei[E + i]; } else { src = dst = i - E; }
    float alpha = p[i] / s[dst];
    atomicAdd(&agg2[2 * dst],     alpha * xl2[2 * src]);
    atomicAdd(&agg2[2 * dst + 1], alpha * xl2[2 * src + 1]);
}

// ---------------------------------------------------------------- bias + log_softmax
__global__ void gat_final(const float* __restrict__ agg2, const float* __restrict__ b2,
                          float* __restrict__ out, int n)
{
    int nd = blockIdx.x * blockDim.x + threadIdx.x;
    if (nd >= n) return;
    float v0 = agg2[2 * nd]     + b2[0];
    float v1 = agg2[2 * nd + 1] + b2[1];
    float mx = fmaxf(v0, v1);
    float lse = mx + __logf(__expf(v0 - mx) + __expf(v1 - mx));
    out[2 * nd]     = v0 - lse;
    out[2 * nd + 1] = v1 - lse;
}

// ---------------------------------------------------------------- launcher
extern "C" void kernel_launch(void* const* d_in, const int* in_sizes, int n_in,
                              void* d_out, int out_size, void* d_ws, size_t ws_size,
                              hipStream_t stream) {
    const float* x    = (const float*)d_in[0];
    const int*   ei   = (const int*)  d_in[1];
    const float* Wl1  = (const float*)d_in[2];
    const float* bl1  = (const float*)d_in[3];
    const float* Wr1  = (const float*)d_in[4];
    const float* br1  = (const float*)d_in[5];
    const float* att1 = (const float*)d_in[6];
    const float* b1   = (const float*)d_in[7];
    const float* Wl2  = (const float*)d_in[8];
    const float* bl2  = (const float*)d_in[9];
    const float* Wr2  = (const float*)d_in[10];
    const float* br2  = (const float*)d_in[11];
    const float* att2 = (const float*)d_in[12];
    const float* b2   = (const float*)d_in[13];

    const int n  = in_sizes[0] / IN_DIM;   // 100000
    const int E  = in_sizes[1] / 2;        // 1600000
    const int EP = E + n;

    // workspace layout (floats)
    float* ws   = (float*)d_ws;
    float* xl1  = ws;                 ws += (size_t)n * HID;
    float* xr1  = ws;                 ws += (size_t)n * HID;
    float* agg1 = ws;                 ws += (size_t)n * HID;
    float* m1   = ws;                 ws += n;
    float* s1   = ws;                 ws += n;
    float* ep1  = ws;                 ws += EP;   // logits, then p
    float* xl2  = ws;                 ws += 2 * n;
    float* xr2  = ws;                 ws += 2 * n;
    float* agg2 = ws;                 ws += 2 * n;
    float* m2   = ws;                 ws += n;
    float* s2   = ws;                 ws += n;
    float* ep2  = ws;                 ws += EP;

    float* out = (float*)d_out;

    const int gN   = (n + TPB - 1) / TPB;
    const int gNH  = (n * HID + TPB - 1) / TPB;
    const int gE   = (EP + TPB - 1) / TPB;
    const int gE4  = (EP * 4 + TPB - 1) / TPB;
    const int gGemm = (n + 127) / 128;   // 8 waves * 16 rows per block

    gat_init_kernel<<<gNH, TPB, 0, stream>>>(m1, s1, agg1, m2, s2, agg2, n);
    gat_gemm1_wmma<<<gGemm, TPB, 0, stream>>>(x, Wl1, bl1, Wr1, br1, xl1, xr1, n);
    gat_logit1<<<gE, TPB, 0, stream>>>(ei, E, n, xl1, xr1, att1, ep1, m1);
    gat_expsum<<<gE, TPB, 0, stream>>>(ei, E, n, m1, ep1, s1);
    gat_agg1<<<gE4, TPB, 0, stream>>>(ei, E, n, ep1, s1, xl1, agg1);
    gat_node_l2<<<gN, TPB, 0, stream>>>(agg1, b1, Wl2, bl2, Wr2, br2, xl2, xr2, n);
    gat_logit2<<<gE, TPB, 0, stream>>>(ei, E, n, xl2, xr2, att2, ep2, m2);
    gat_expsum<<<gE, TPB, 0, stream>>>(ei, E, n, m2, ep2, s2);
    gat_agg2<<<gE, TPB, 0, stream>>>(ei, E, n, ep2, s2, xl2, agg2);
    gat_final<<<gN, TPB, 0, stream>>>(agg2, b2, out, n);
}
